// MultiHeadSelfAttention_31911607009680
// MI455X (gfx1250) — matrix operand order repaired and checked by reading, not compiled, not run on hardware
//
#include <hip/hip_runtime.h>
#include <hip/hip_bf16.h>
#include <stdint.h>

// ---------------------------------------------------------------------------
// MHA forward for MI455X (gfx1250, wave32, WMMA, bf16 compute / f32 accum).
// B=2, T=2048, D_MODEL=1024, N_HEAD=16, D_HEAD=64.
//   1) x f32->bf16; W_qkv/W_proj f32 -> transposed bf16 (one-time)
//   2) QKV GEMM: 64x128 block tile, 32x32 wave tile, K-step 64,
//      double-buffered LDS staged with global_load_async_to_lds_b128
//      (ASYNCcnt pipeline: issue next tile, wait on previous, compute)
//   3) V pre-transpose per head -> Vt[b*h][d][t]
//   4) Flash attention: fragments straight from global (L2-resident),
//      no barriers, unmasked main loop + masked diagonal tile,
//      P reshaped through per-wave LDS
//   5) proj GEMM -> f32 out
// ---------------------------------------------------------------------------

typedef __attribute__((ext_vector_type(16))) __bf16 v16bf;
typedef __attribute__((ext_vector_type(8)))  __bf16 v8bf;
typedef __attribute__((ext_vector_type(8)))  float  v8f;
typedef __attribute__((ext_vector_type(4)))  int    v4i;

union FragBF {
    v16bf v;
    v8bf  h[2];
};

// --------------------------- async copy to LDS ------------------------------
#if __has_builtin(__builtin_amdgcn_global_load_async_to_lds_b128)
#define HAS_ASYNC_LDS 1
#else
#define HAS_ASYNC_LDS 0
#endif

__device__ __forceinline__ void async_copy16(const __bf16* __restrict__ g,
                                             __bf16* l) {
#if HAS_ASYNC_LDS
    __builtin_amdgcn_global_load_async_to_lds_b128(
        (__attribute__((address_space(1))) v4i*)(uintptr_t)g,
        (__attribute__((address_space(3))) v4i*)(uint32_t)(uintptr_t)l,
        0, 0);
#else
    *(v8bf*)l = *(const v8bf*)g;
#endif
}

template <int N>
__device__ __forceinline__ void async_join() {
#if HAS_ASYNC_LDS
#if __has_builtin(__builtin_amdgcn_s_wait_asynccnt)
    __builtin_amdgcn_s_wait_asynccnt(N);
#else
    asm volatile("s_wait_asynccnt %0" ::"i"(N) : "memory");
#endif
#endif
}

// ---------------------------------------------------------------------------
// f32 -> bf16 elementwise convert
// ---------------------------------------------------------------------------
__global__ void cvt_f32_to_bf16(const float* __restrict__ in,
                                __bf16* __restrict__ out, int n) {
    int i = blockIdx.x * blockDim.x + threadIdx.x;
    if (i < n) out[i] = (__bf16)in[i];
}

// ---------------------------------------------------------------------------
// Transpose+convert: in[R][C] f32 -> out[C][R] bf16. Block (32,8), tile 32x32.
// ---------------------------------------------------------------------------
__global__ void transpose_cvt_f32_bf16(const float* __restrict__ in,
                                       __bf16* __restrict__ out,
                                       int R, int C) {
    __shared__ float tile[32][33];
    const int c0 = blockIdx.x * 32, r0 = blockIdx.y * 32;
    const int tx = threadIdx.x;
    for (int i = threadIdx.y; i < 32; i += 8)
        tile[i][tx] = in[(size_t)(r0 + i) * C + c0 + tx];
    __syncthreads();
    for (int i = threadIdx.y; i < 32; i += 8)
        out[(size_t)(c0 + i) * R + r0 + tx] = (__bf16)tile[tx][i];
}

// ---------------------------------------------------------------------------
// Per-head V transpose: qkv V slice [t][d] bf16 -> vt[bh][d][t] bf16.
// Grid (T/32, 64/32, B*H), block (32,8).
// ---------------------------------------------------------------------------
__global__ void transpose_v_bf16(const __bf16* __restrict__ qkv,
                                 __bf16* __restrict__ vt) {
    __shared__ __bf16 tile[32][33];
    const int bh = blockIdx.z, b = bh >> 4, h = bh & 15;
    const int t0 = blockIdx.x * 32, d0 = blockIdx.y * 32;
    const int tx = threadIdx.x;
    const __bf16* src = qkv + (size_t)b * 2048 * 3072 + 2048 + h * 64;
    for (int i = threadIdx.y; i < 32; i += 8)
        tile[i][tx] = src[(size_t)(t0 + i) * 3072 + d0 + tx];
    __syncthreads();
    __bf16* dst = vt + (size_t)bh * 64 * 2048;
    for (int i = threadIdx.y; i < 32; i += 8)
        dst[(size_t)(d0 + i) * 2048 + t0 + tx] = tile[tx][i];
}

// ---------------------------------------------------------------------------
// GEMM: C[M,N] = A[M,K] * Bt[N,K]^T + bias[N]
// A row-major bf16, Bt = B^T row-major bf16 (all LDS stores vectorized).
// Block 256 = 8 waves; block tile 64(M)x128(N); wave tile 32x32;
// K-step 64 (2 chunks, 8 WMMAs/wave/stage); double-buffered async staging.
// M mult of 64, N mult of 128, K mult of 64.
// ---------------------------------------------------------------------------
constexpr int AP = 72;  // LDS row pitch (elements), mult of 8, bank-skewed

template <bool OUT_F32>
__global__ __launch_bounds__(256)
void gemm_bf16_wmma(const __bf16* __restrict__ A,
                    const __bf16* __restrict__ Bt,
                    const float* __restrict__ bias,
                    float* __restrict__ Cf,
                    __bf16* __restrict__ Cb,
                    int M, int N, int K) {
    __shared__ __bf16 As[2][64 * AP];    // [buf][m][k]
    __shared__ __bf16 Bs[2][128 * AP];   // [buf][n][k]

    const int tid  = threadIdx.x;
    const int lane = tid & 31;
    const int wave = tid >> 5;
    const int half = lane >> 4;
    const int l16  = lane & 15;

    const int m0 = blockIdx.y * 64;
    const int n0 = blockIdx.x * 128;
    const int wm = (wave & 1) * 32;   // 2 m-positions
    const int wn = (wave >> 1) * 32;  // 4 n-positions

    v8f acc[2][2] = {{{}, {}}, {{}, {}}};

    // staging assignments: A tile 64x64 (2 copies/thread), B tile 128x64 (4)
    const int arow = tid >> 2, acg = (tid & 3) * 16;
    const int brow = tid >> 1, bcg = (tid & 1) * 32;

    auto stage = [&](int k0, int buf) {
        const __bf16* ag = A + (size_t)(m0 + arow) * K + k0 + acg;
        __bf16* al = &As[buf][arow * AP + acg];
        async_copy16(ag, al);
        async_copy16(ag + 8, al + 8);
        const __bf16* bg = Bt + (size_t)(n0 + brow) * K + k0 + bcg;
        __bf16* bl = &Bs[buf][brow * AP + bcg];
        async_copy16(bg, bl);
        async_copy16(bg + 8, bl + 8);
        async_copy16(bg + 16, bl + 16);
        async_copy16(bg + 24, bl + 24);
    };

    stage(0, 0);  // prologue: 6 async copies in flight

    for (int k0 = 0, it = 0; k0 < K; k0 += 64, ++it) {
        const int buf = it & 1;
        if (k0 + 64 < K) {
            stage(k0 + 64, buf ^ 1);  // 12 in flight
            async_join<6>();          // oldest 6 (this stage) have landed
        } else {
            async_join<0>();
        }
        __syncthreads();

#pragma unroll
        for (int c = 0; c < 2; ++c) {
            FragBF a[2], b[2];
#pragma unroll
            for (int i = 0; i < 2; ++i) {
                const __bf16* ap = &As[buf][(wm + i * 16 + l16) * AP + c * 32];
                a[i].h[0] = *(const v8bf*)(ap + half * 16);
                a[i].h[1] = *(const v8bf*)(ap + half * 16 + 8);
                const __bf16* bp =
                    &Bs[buf][(wn + i * 16 + l16) * AP + c * 32 + half * 16];
                b[i].h[0] = *(const v8bf*)(bp);
                b[i].h[1] = *(const v8bf*)(bp + 8);
            }
#pragma unroll
            for (int i = 0; i < 2; ++i)
#pragma unroll
                for (int j = 0; j < 2; ++j)
                    acc[i][j] = __builtin_amdgcn_wmma_f32_16x16x32_bf16(
                        false, a[i].v, false, b[j].v, (short)0, acc[i][j],
                        false, false);
        }
        __syncthreads();  // all waves done reading buf before it is re-staged
    }

    // epilogue: D elem r -> row (i*16 + r + 8*half), col (j*16 + l16)
#pragma unroll
    for (int j = 0; j < 2; ++j) {
        const float bj = bias[n0 + wn + j * 16 + l16];
#pragma unroll
        for (int i = 0; i < 2; ++i) {
#pragma unroll
            for (int r = 0; r < 8; ++r) {
                const int row = m0 + wm + i * 16 + r + 8 * half;
                const int col = n0 + wn + j * 16 + l16;
                const float v = acc[i][j][r] + bj;
                if (OUT_F32) Cf[(size_t)row * N + col] = v;
                else         Cb[(size_t)row * N + col] = (__bf16)v;
            }
        }
    }
}

// ---------------------------------------------------------------------------
// Flash attention (causal). qkv: [B,T,3072] bf16 (Q@0, K@1024, V@2048),
// vt: [B*H][64][2048] bf16, ctx: [B,T,1024] bf16.
// Grid (B*H, T/128), block 256 = 8 independent waves, 16 query rows each.
// Unmasked main loop; only the diagonal tile pays for masking.
// ---------------------------------------------------------------------------
constexpr int PSP = 72;  // P staging pitch

__global__ __launch_bounds__(256)
void flash_attn_wmma(const __bf16* __restrict__ qkv,
                     const __bf16* __restrict__ vt,
                     __bf16* __restrict__ ctx) {
    __shared__ __bf16 Ps[8][16 * PSP];

    const int tid  = threadIdx.x;
    const int lane = tid & 31;
    const int wave = tid >> 5;
    const int half = lane >> 4;
    const int l16  = lane & 15;

    const int bh = blockIdx.x, b = bh >> 4, h = bh & 15;
    const int q0 = blockIdx.y * 128 + wave * 16;  // wave's 16 query rows

    const __bf16* qb  = qkv + (size_t)b * 2048 * 3072 + h * 64;
    const __bf16* kbp = qb + 1024;
    const __bf16* vtb = vt + (size_t)bh * 64 * 2048;

    // Q fragments (A layout): row = l16, contiguous 16B runs in global
    FragBF qa[2];
#pragma unroll
    for (int c = 0; c < 2; ++c) {
        const __bf16* qp = qb + (size_t)(q0 + l16) * 3072 + c * 32;
        qa[c].h[0] = *(const v8bf*)(qp + half * 16);
        qa[c].h[1] = *(const v8bf*)(qp + half * 16 + 8);
    }

    float mrow[8], lrow[8];
    v8f oacc[4] = {{}, {}, {}, {}};
#pragma unroll
    for (int r = 0; r < 8; ++r) { mrow[r] = -1e30f; lrow[r] = 0.0f; }

    auto process_tile = [&](int kb, bool masked) {
        // --- S = Q K^T : B-frags straight from global K (contiguous) ---
        v8f s[4];
#pragma unroll
        for (int nt = 0; nt < 4; ++nt) {
            s[nt] = (v8f){};
#pragma unroll
            for (int c = 0; c < 2; ++c) {
                FragBF bk;
                const __bf16* kp = kbp + (size_t)(kb + nt * 16 + l16) * 3072 +
                                   c * 32 + half * 16;
                bk.h[0] = *(const v8bf*)(kp);
                bk.h[1] = *(const v8bf*)(kp + 8);
                s[nt] = __builtin_amdgcn_wmma_f32_16x16x32_bf16(
                    false, qa[c].v, false, bk.v, (short)0, s[nt], false, false);
            }
        }

        // --- scale (+ mask on diagonal tile) + online softmax; store P ---
#pragma unroll
        for (int r = 0; r < 8; ++r) {
            const int qg = q0 + r + 8 * half;
            float p[4];
#pragma unroll
            for (int nt = 0; nt < 4; ++nt) {
                float v = s[nt][r] * 0.125f;
                if (masked && (kb + nt * 16 + l16 > qg)) v = -1e30f;
                p[nt] = v;
            }
            float mx = fmaxf(fmaxf(p[0], p[1]), fmaxf(p[2], p[3]));
            mx = fmaxf(mx, __shfl_xor(mx, 1, 32));
            mx = fmaxf(mx, __shfl_xor(mx, 2, 32));
            mx = fmaxf(mx, __shfl_xor(mx, 4, 32));
            mx = fmaxf(mx, __shfl_xor(mx, 8, 32));
            const float mnew = fmaxf(mrow[r], mx);
            float rs = 0.0f;
#pragma unroll
            for (int nt = 0; nt < 4; ++nt) {
                p[nt] = __expf(p[nt] - mnew);
                rs += p[nt];
            }
            rs += __shfl_xor(rs, 1, 32);
            rs += __shfl_xor(rs, 2, 32);
            rs += __shfl_xor(rs, 4, 32);
            rs += __shfl_xor(rs, 8, 32);
            const float alpha = __expf(mrow[r] - mnew);
            lrow[r] = lrow[r] * alpha + rs;
            mrow[r] = mnew;
#pragma unroll
            for (int dt = 0; dt < 4; ++dt) oacc[dt][r] *= alpha;
#pragma unroll
            for (int nt = 0; nt < 4; ++nt)
                Ps[wave][(r + 8 * half) * PSP + nt * 16 + l16] = (__bf16)p[nt];
        }

        // --- O += P V : A-frag from per-wave LDS (in-order DS, no barrier),
        //                B-frag straight from global Vt (contiguous) ---
#pragma unroll
        for (int c = 0; c < 2; ++c) {
            FragBF pa;
            const __bf16* pp = &Ps[wave][l16 * PSP + c * 32];
            pa.h[0] = *(const v8bf*)(pp + half * 16);
            pa.h[1] = *(const v8bf*)(pp + half * 16 + 8);
#pragma unroll
            for (int dt = 0; dt < 4; ++dt) {
                FragBF bv;
                const __bf16* vp = vtb + (size_t)(dt * 16 + l16) * 2048 + kb +
                                   c * 32 + half * 16;
                bv.h[0] = *(const v8bf*)(vp);
                bv.h[1] = *(const v8bf*)(vp + 8);
                oacc[dt] = __builtin_amdgcn_wmma_f32_16x16x32_bf16(
                    false, pa.v, false, bv.v, (short)0, oacc[dt], false, false);
            }
        }
    };

    const int nkt = q0 / 64 + 1;  // causal bound for this wave
    for (int kt = 0; kt < nkt - 1; ++kt) process_tile(kt * 64, false);
    process_tile((nkt - 1) * 64, true);  // diagonal tile pays for masking

    // --- normalize + store ctx bf16 ---
#pragma unroll
    for (int dt = 0; dt < 4; ++dt) {
#pragma unroll
        for (int r = 0; r < 8; ++r) {
            const int qg = q0 + r + 8 * half;
            const float o = oacc[dt][r] / lrow[r];
            ctx[((size_t)(b * 2048 + qg)) * 1024 + h * 64 + dt * 16 + l16] =
                (__bf16)o;
        }
    }
}

// ---------------------------------------------------------------------------
// Host-side launch
// ---------------------------------------------------------------------------
extern "C" void kernel_launch(void* const* d_in, const int* in_sizes, int n_in,
                              void* d_out, int out_size, void* d_ws, size_t ws_size,
                              hipStream_t stream) {
    (void)in_sizes; (void)n_in; (void)out_size; (void)ws_size;

    const float* x     = (const float*)d_in[0];  // [2,2048,1024]
    const float* Wqkv  = (const float*)d_in[1];  // [1024,3072]
    const float* bqkv  = (const float*)d_in[2];  // [3072]
    const float* Wproj = (const float*)d_in[3];  // [1024,1024]
    const float* bproj = (const float*)d_in[4];  // [1024]
    float* out = (float*)d_out;                  // [2,2048,1024]

    char* ws = (char*)d_ws;
    __bf16* x_bf      = (__bf16*)(ws);                 //  8 MiB [4096][1024]
    __bf16* wqkvT_bf  = (__bf16*)(ws + (8u  << 20));   //  6 MiB [3072][1024]
    __bf16* wprojT_bf = (__bf16*)(ws + (14u << 20));   //  2 MiB [1024][1024]
    __bf16* qkv_bf    = (__bf16*)(ws + (16u << 20));   // 24 MiB [4096][3072]
    __bf16* vt_bf     = (__bf16*)(ws + (40u << 20));   //  8 MiB [32][64][2048]
    __bf16* ctx_bf    = (__bf16*)(ws + (48u << 20));   //  8 MiB [4096][1024]

    const int NX = 2 * 2048 * 1024;

    cvt_f32_to_bf16<<<(NX + 255) / 256, 256, 0, stream>>>(x, x_bf, NX);
    transpose_cvt_f32_bf16<<<dim3(3072 / 32, 1024 / 32), dim3(32, 8), 0, stream>>>(
        Wqkv, wqkvT_bf, 1024, 3072);
    transpose_cvt_f32_bf16<<<dim3(1024 / 32, 1024 / 32), dim3(32, 8), 0, stream>>>(
        Wproj, wprojT_bf, 1024, 1024);

    // qkv = x @ W_qkv + b_qkv   (M=4096, N=3072, K=1024) -> bf16
    gemm_bf16_wmma<false><<<dim3(3072 / 128, 4096 / 64), 256, 0, stream>>>(
        x_bf, wqkvT_bf, bqkv, nullptr, qkv_bf, 4096, 3072, 1024);

    // Vt per head
    transpose_v_bf16<<<dim3(2048 / 32, 2, 32), dim3(32, 8), 0, stream>>>(
        qkv_bf, vt_bf);

    // causal flash attention -> ctx bf16
    flash_attn_wmma<<<dim3(32, 2048 / 128), 256, 0, stream>>>(
        qkv_bf, vt_bf, ctx_bf);

    // out = ctx @ W_proj + b_proj   (M=4096, N=1024, K=1024) -> f32
    gemm_bf16_wmma<true><<<dim3(1024 / 128, 4096 / 64), 256, 0, stream>>>(
        ctx_bf, wprojT_bf, bproj, out, nullptr, 4096, 1024, 1024);
}
